// QCNetOEMapEncoder_83021717832157
// MI455X (gfx1250) — compile-verified, matrix operand order checked by reading, not run on hardware
//
#include <hip/hip_runtime.h>

// ---------------- problem constants ----------------
constexpr int cB = 2, cPL = 512, cPT = 128, cK = 32;
constexpr int cH = 128, cF = 64;
constexpr long N_PT  = (long)cB * cPL * cPT;  // 131072
constexpr long N_PLK = (long)cB * cPL * cK;   // 32768
constexpr long N_PL  = (long)cB * cPL;        // 1024

// packed-weight block sizes (in halves): [ntiles][kchunks][32 lanes][16 halves]
constexpr int W1H  = 8 * 5 * 512;    // 128x129 (K padded to 160)
constexpr int W2H  = 8 * 4 * 512;    // 128x128
constexpr int WGH  = 8 * 8 * 512;    // 128x256
constexpr int WF1H = 32 * 4 * 512;   // 512x128
constexpr int WF2H = 8 * 16 * 512;   // 128x512

typedef __attribute__((ext_vector_type(16))) _Float16 v16h;
typedef __attribute__((ext_vector_type(8)))  float    v8f;

// ---------------- WMMA helpers ----------------
__device__ __forceinline__ v8f wmma_f16(v16h a, v16h b, v8f c) {
  return __builtin_amdgcn_wmma_f32_16x16x32_f16(false, a, false, b, (short)0, c,
                                                false, false);
}

// A fragment (16x32 f16) from an LDS row-major tile, row stride ldr halves.
__device__ __forceinline__ v16h load_a_frag(const _Float16* src, int ldr, int kb) {
  const int lane = threadIdx.x & 31;
  const int m = lane & 15;
  const int c0 = kb + ((lane & 16) ? 8 : 0);
  const _Float16* row = src + m * ldr + c0;
  v16h a;
#pragma unroll
  for (int j = 0; j < 8; ++j) a[j] = row[j];
#pragma unroll
  for (int j = 0; j < 8; ++j) a[8 + j] = row[16 + j];
  return a;
}

// B fragment from pre-packed global memory: lane-contiguous 16 halves.
__device__ __forceinline__ v16h load_b_frag(const _Float16* p) {
  const int lane = threadIdx.x & 31;
  return *(const v16h*)(p + lane * 16);
}

// C/D layout: VGPR r -> row = r + (lane>=16 ? 8 : 0), col = nb + (lane&15)
__device__ __forceinline__ void store_c(float* dst, int ld, int nb, v8f acc) {
  const int lane = threadIdx.x & 31;
  const int n = nb + (lane & 15);
  const int mh = (lane & 16) ? 8 : 0;
#pragma unroll
  for (int r = 0; r < 8; ++r) dst[(mh + r) * ld + n] = acc[r];
}

// block-wide LayerNorm over Rx128 f32 LDS tile -> f16 LDS tile (256 threads).
// Optional f32 copy of the LN output (pre-relu) to global at (e0+row)*128+col.
template <int R>
__device__ __forceinline__ void ln_block(const float* __restrict__ src,
                                         _Float16* __restrict__ dst, bool relu,
                                         float* __restrict__ gdst, long e0) {
  constexpr int TPR = 256 / R;   // threads per row (8 or 16)
  constexpr int CPT = cH / TPR;  // cols per thread (16 or 8)
  const int t = threadIdx.x;
  const int r = t / TPR, cg = t % TPR;
  const float* row = src + r * cH + cg * CPT;
  float v[CPT];
  float s = 0.f;
#pragma unroll
  for (int j = 0; j < CPT; ++j) { v[j] = row[j]; s += v[j]; }
#pragma unroll
  for (int m = TPR / 2; m >= 1; m >>= 1) s += __shfl_xor(s, m);
  const float mean = s * (1.f / cH);
  float s2 = 0.f;
#pragma unroll
  for (int j = 0; j < CPT; ++j) { float d = v[j] - mean; s2 += d * d; }
#pragma unroll
  for (int m = TPR / 2; m >= 1; m >>= 1) s2 += __shfl_xor(s2, m);
  const float inv = rsqrtf(s2 * (1.f / cH) + 1e-5f);
  _Float16* drow = dst + r * cH + cg * CPT;
#pragma unroll
  for (int j = 0; j < CPT; ++j) {
    float o = (v[j] - mean) * inv;
    if (gdst) gdst[(e0 + r) * cH + cg * CPT + j] = o;
    if (relu) o = fmaxf(o, 0.f);
    drow[j] = (_Float16)o;
  }
}

// ---------------- async global -> LDS tile copy (CDNA5 TDM-lite path) ----
// Moves nfloats f32 (multiple of 4*256) from gsrc to the LDS buffer ldst using
// GLOBAL_LOAD_ASYNC_TO_LDS_B128 (tracked by ASYNCcnt). Flat shared addresses
// map to LDS offsets via low-32-bit truncation (ISA 10.2 aperture rule).
__device__ __forceinline__ void async_tile_load(const float* __restrict__ gsrc,
                                                float* __restrict__ ldst,
                                                int nfloats) {
  const int t = threadIdx.x;
  const int chunks = nfloats >> 2;  // 16B chunks
  for (int c = t; c < chunks; c += 256) {
    const unsigned lds_off = (unsigned)(uintptr_t)(ldst + c * 4);
    const float* ga = gsrc + c * 4;
    asm volatile("global_load_async_to_lds_b128 %0, %1, off"
                 :
                 : "v"(lds_off), "v"(ga)
                 : "memory");
  }
  asm volatile("s_wait_asynccnt 0" ::: "memory");
}

// ---------------- weight packer: f32 (N,K) row-major -> f16 WMMA-B fragments
__global__ void pack_b_kernel(_Float16* __restrict__ dst,
                              const float* __restrict__ src, int N, int Kdim,
                              int KC) {
  const int kc = blockIdx.x, nt = blockIdx.y, lane = threadIdx.x;
  const int nn = nt * 16 + (lane & 15);
  const int base = (lane & 16) ? 8 : 0;
  _Float16* d = dst + (((long)nt * KC + kc) * 32 + lane) * 16;
#pragma unroll
  for (int j = 0; j < 16; ++j) {
    const int v = j >> 1;
    const int kk =
        kc * 32 + base + ((v & 3) * 2 + (j & 1)) + ((v >= 4) ? 16 : 0);
    float val = (kk < Kdim) ? src[(long)nn * Kdim + kk] : 0.f;
    d[j] = (_Float16)val;
  }
}

// ---------------- Fourier MLP stage (32-row tiles, B-frag reuse x2) --------
// out[e,:] = relu(LN( sum_d relu(LN(feat_d W1_d^T + b1_d)) W2_d^T + b2_d + cat )) Wo^T + bo
__global__ __launch_bounds__(256) void fourier_kernel(
    const float* __restrict__ xs, long dimStride, int ND,
    const float* __restrict__ freqs, const _Float16* __restrict__ w1p,
    const float* __restrict__ b1, const _Float16* __restrict__ w2p,
    const float* __restrict__ b2, const _Float16* __restrict__ wop,
    const float* __restrict__ bo, int catMode, const int* __restrict__ idx0,
    const int* __restrict__ idx1, const float* __restrict__ emb0,
    const float* __restrict__ emb1, float* __restrict__ out) {
  __shared__ __align__(32) _Float16 featH[32 * 160];
  __shared__ __align__(32) float fbuf[32 * 128];
  __shared__ __align__(32) _Float16 hbuf16[32 * 128];
  __shared__ float xsv[32];
  const int t = threadIdx.x;
  const int lane = t & 31;
  const int w = t >> 5;
  const long e0 = (long)blockIdx.x * 32;
  const int n = w * 16 + (lane & 15);
  const int mh = (lane & 16) ? 8 : 0;

  v8f hacc0, hacc1;
#pragma unroll
  for (int r = 0; r < 8; ++r) { hacc0[r] = 0.f; hacc1[r] = 0.f; }

  for (int d = 0; d < ND; ++d) {
    if (t < 32) xsv[t] = xs[(long)d * dimStride + e0 + t];
    __syncthreads();
    for (int i = t; i < 32 * 160; i += 256) {
      const int row = i / 160, col = i % 160;
      const float x = xsv[row];
      float val;
      if (col < 64)
        val = cosf(x * freqs[d * cF + col] * 6.28318530717958647692f);
      else if (col < 128)
        val = sinf(x * freqs[d * cF + col - 64] * 6.28318530717958647692f);
      else if (col == 128)
        val = x;
      else
        val = 0.f;
      featH[row * 160 + col] = (_Float16)val;
    }
    __syncthreads();
    {  // GEMM1: feat(32x160) x W1 -> 32x128 (+b1)
      v8f a0, a1;
      const float bb = b1[d * cH + n];
#pragma unroll
      for (int r = 0; r < 8; ++r) { a0[r] = bb; a1[r] = bb; }
      const _Float16* wp = w1p + (long)d * W1H + (long)w * (5 * 512);
#pragma unroll
      for (int kc = 0; kc < 5; ++kc) {
        v16h bf = load_b_frag(wp + kc * 512);
        a0 = wmma_f16(load_a_frag(featH, 160, kc * 32), bf, a0);
        a1 = wmma_f16(load_a_frag(featH + 16 * 160, 160, kc * 32), bf, a1);
      }
      store_c(fbuf, cH, w * 16, a0);
      store_c(fbuf + 16 * cH, cH, w * 16, a1);
    }
    __syncthreads();
    ln_block<32>(fbuf, hbuf16, true, nullptr, 0);
    __syncthreads();
    {  // GEMM2 accumulate across dims
      const _Float16* wp = w2p + (long)d * W2H + (long)w * (4 * 512);
#pragma unroll
      for (int kc = 0; kc < 4; ++kc) {
        v16h bf = load_b_frag(wp + kc * 512);
        hacc0 = wmma_f16(load_a_frag(hbuf16, cH, kc * 32), bf, hacc0);
        hacc1 = wmma_f16(load_a_frag(hbuf16 + 16 * cH, cH, kc * 32), bf, hacc1);
      }
    }
    __syncthreads();
  }
  {  // add sum of b2 and stage hsum
    float bs = 0.f;
    for (int d = 0; d < ND; ++d) bs += b2[d * cH + n];
#pragma unroll
    for (int r = 0; r < 8; ++r) { hacc0[r] += bs; hacc1[r] += bs; }
    store_c(fbuf, cH, w * 16, hacc0);
    store_c(fbuf + 16 * cH, cH, w * 16, hacc1);
  }
  __syncthreads();
  if (catMode == 1) {
    for (int i = t; i < 32 * 128; i += 256) {
      const int row = i >> 7, col = i & 127;
      const long e = e0 + row;
      fbuf[i] += emb0[(long)idx0[e] * cH + col] + emb1[(long)idx1[e] * cH + col];
    }
    __syncthreads();
  } else if (catMode == 2) {
    for (int i = t; i < 32 * 128; i += 256) {
      const int row = i >> 7, col = i & 127;
      fbuf[i] += emb0[(long)idx0[e0 + row] * cH + col];
    }
    __syncthreads();
  }
  ln_block<32>(fbuf, hbuf16, true, nullptr, 0);
  __syncthreads();
  {  // GEMM3: relu(ln(h)) x Wo -> out (+bo)
    v8f a0, a1;
    const float bb = bo[n];
#pragma unroll
    for (int r = 0; r < 8; ++r) { a0[r] = bb; a1[r] = bb; }
    const _Float16* wp = wop + (long)w * (4 * 512);
#pragma unroll
    for (int kc = 0; kc < 4; ++kc) {
      v16h bf = load_b_frag(wp + kc * 512);
      a0 = wmma_f16(load_a_frag(hbuf16, cH, kc * 32), bf, a0);
      a1 = wmma_f16(load_a_frag(hbuf16 + 16 * cH, cH, kc * 32), bf, a1);
    }
#pragma unroll
    for (int r = 0; r < 8; ++r) {
      out[(e0 + mh + r) * cH + n] = a0[r];
      out[(e0 + 16 + mh + r) * cH + n] = a1[r];
    }
  }
}

// ---------------- attention k/v projection (32-row tiles) ------------------
// k = ln(xs) Wk^T + ln(r) Wkr^T ; v = ln(xs) Wv^T + bv + ln(r) Wvr^T + bvr
__global__ __launch_bounds__(256) void kv_kernel(
    const float* __restrict__ xsrc, const float* __restrict__ rsrc,
    const _Float16* __restrict__ wkp, const _Float16* __restrict__ wkrp,
    const _Float16* __restrict__ wvp, const _Float16* __restrict__ wvrp,
    const float* __restrict__ bv, const float* __restrict__ bvr,
    _Float16* __restrict__ kout, _Float16* __restrict__ vout) {
  __shared__ __align__(32) float fbuf[32 * 128];
  __shared__ __align__(32) _Float16 xs16[32 * 128];
  __shared__ __align__(32) _Float16 rn16[32 * 128];
  const int t = threadIdx.x;
  const int lane = t & 31;
  const int w = t >> 5;
  const long e0 = (long)blockIdx.x * 32;
  const int n = w * 16 + (lane & 15);
  const int mh = (lane & 16) ? 8 : 0;

  async_tile_load(xsrc + e0 * cH, fbuf, 32 * cH);
  __syncthreads();
  ln_block<32>(fbuf, xs16, false, nullptr, 0);
  __syncthreads();
  async_tile_load(rsrc + e0 * cH, fbuf, 32 * cH);
  __syncthreads();
  ln_block<32>(fbuf, rn16, false, nullptr, 0);
  __syncthreads();
  {  // k
    v8f a0, a1;
#pragma unroll
    for (int r = 0; r < 8; ++r) { a0[r] = 0.f; a1[r] = 0.f; }
#pragma unroll
    for (int kc = 0; kc < 4; ++kc) {
      v16h bf = load_b_frag(wkp + (long)w * 2048 + kc * 512);
      a0 = wmma_f16(load_a_frag(xs16, cH, kc * 32), bf, a0);
      a1 = wmma_f16(load_a_frag(xs16 + 16 * cH, cH, kc * 32), bf, a1);
    }
#pragma unroll
    for (int kc = 0; kc < 4; ++kc) {
      v16h bf = load_b_frag(wkrp + (long)w * 2048 + kc * 512);
      a0 = wmma_f16(load_a_frag(rn16, cH, kc * 32), bf, a0);
      a1 = wmma_f16(load_a_frag(rn16 + 16 * cH, cH, kc * 32), bf, a1);
    }
#pragma unroll
    for (int r = 0; r < 8; ++r) {
      kout[(e0 + mh + r) * cH + n] = (_Float16)a0[r];
      kout[(e0 + 16 + mh + r) * cH + n] = (_Float16)a1[r];
    }
  }
  {  // v
    v8f a0, a1;
    const float bb = bv[n] + bvr[n];
#pragma unroll
    for (int r = 0; r < 8; ++r) { a0[r] = bb; a1[r] = bb; }
#pragma unroll
    for (int kc = 0; kc < 4; ++kc) {
      v16h bf = load_b_frag(wvp + (long)w * 2048 + kc * 512);
      a0 = wmma_f16(load_a_frag(xs16, cH, kc * 32), bf, a0);
      a1 = wmma_f16(load_a_frag(xs16 + 16 * cH, cH, kc * 32), bf, a1);
    }
#pragma unroll
    for (int kc = 0; kc < 4; ++kc) {
      v16h bf = load_b_frag(wvrp + (long)w * 2048 + kc * 512);
      a0 = wmma_f16(load_a_frag(rn16, cH, kc * 32), bf, a0);
      a1 = wmma_f16(load_a_frag(rn16 + 16 * cH, cH, kc * 32), bf, a1);
    }
#pragma unroll
    for (int r = 0; r < 8; ++r) {
      vout[(e0 + mh + r) * cH + n] = (_Float16)a0[r];
      vout[(e0 + 16 + mh + r) * cH + n] = (_Float16)a1[r];
    }
  }
}

// ---------------- q / s-vec projection of dst rows ----------------
__global__ __launch_bounds__(256) void qs_kernel(
    const float* __restrict__ xdst, const _Float16* __restrict__ wqp,
    const float* __restrict__ bq, const _Float16* __restrict__ wsp,
    const float* __restrict__ bs, float* __restrict__ xdln,
    float* __restrict__ qout, float* __restrict__ svout) {
  __shared__ __align__(32) float fbuf[16 * 128];
  __shared__ __align__(32) _Float16 x16[16 * 128];
  const int t = threadIdx.x;
  const int lane = t & 31;
  const int w = t >> 5;
  const long e0 = (long)blockIdx.x * 16;
  const int n = w * 16 + (lane & 15);
  const int mh = (lane & 16) ? 8 : 0;

  async_tile_load(xdst + e0 * cH, fbuf, 16 * cH);
  __syncthreads();
  ln_block<16>(fbuf, x16, false, xdln, e0);
  __syncthreads();
  {  // q
    v8f acc;
    const float bb = bq[n];
#pragma unroll
    for (int r = 0; r < 8; ++r) acc[r] = bb;
#pragma unroll
    for (int kc = 0; kc < 4; ++kc)
      acc = wmma_f16(load_a_frag(x16, cH, kc * 32),
                     load_b_frag(wqp + (long)w * 2048 + kc * 512), acc);
#pragma unroll
    for (int r = 0; r < 8; ++r) qout[(e0 + mh + r) * cH + n] = acc[r];
  }
  {  // svec = ln(xd) Ws^T + Bs
    v8f acc;
    const float bb = bs[n];
#pragma unroll
    for (int r = 0; r < 8; ++r) acc[r] = bb;
#pragma unroll
    for (int kc = 0; kc < 4; ++kc)
      acc = wmma_f16(load_a_frag(x16, cH, kc * 32),
                     load_b_frag(wsp + (long)w * 2048 + kc * 512), acc);
#pragma unroll
    for (int r = 0; r < 8; ++r) svout[(e0 + mh + r) * cH + n] = acc[r];
  }
}

// ---------------- attention core: sim -> softmax -> agg ----------------
__global__ __launch_bounds__(256) void attn_core_kernel(
    const float* __restrict__ q, const _Float16* __restrict__ k,
    const _Float16* __restrict__ v, const int* __restrict__ mask, int S,
    float* __restrict__ agg) {
  __shared__ float sim[128 * 8];
  __shared__ float qsh[128];
  __shared__ float mx[8], sm[8];
  const int bp = blockIdx.x;
  const int t = threadIdx.x;
  if (t < 128) qsh[t] = q[(long)bp * cH + t];
  __syncthreads();
  const _Float16* kb = k + (long)bp * S * cH;
  for (int i = t; i < S * 8; i += 256) {
    const int s = i >> 3, h = i & 7;
    const _Float16* kr = kb + s * cH + h * 16;
    float acc = 0.f;
#pragma unroll
    for (int d2 = 0; d2 < 16; ++d2) acc += qsh[h * 16 + d2] * (float)kr[d2];
    sim[i] = mask[(long)bp * S + s] ? acc * 0.25f : -1e9f;
  }
  __syncthreads();
  if (t < 8) {
    float m = -3.0e38f;
    for (int s = 0; s < S; ++s) m = fmaxf(m, sim[s * 8 + t]);
    float ssum = 0.f;
    for (int s = 0; s < S; ++s) ssum += expf(sim[s * 8 + t] - m);
    mx[t] = m;
    sm[t] = 1.f / ssum;
  }
  __syncthreads();
  for (int i = t; i < S * 8; i += 256) {
    const int h = i & 7;
    sim[i] = expf(sim[i] - mx[h]) * sm[h];
  }
  __syncthreads();
  const _Float16* vb = v + (long)bp * S * cH;
  if (t < 128) {
    const int h = t >> 4;
    float acc = 0.f;
    for (int s = 0; s < S; ++s) acc += sim[s * 8 + h] * (float)vb[s * cH + t];
    agg[(long)bp * cH + t] = acc;
  }
}

// ---------------- gating + out-proj + FFN ----------------
__global__ __launch_bounds__(256) void post_kernel(
    const float* __restrict__ agg, const float* __restrict__ xdln,
    const float* __restrict__ sv, const float* __restrict__ xdst,
    const _Float16* __restrict__ wgp, const float* __restrict__ bg,
    const _Float16* __restrict__ wop2, const float* __restrict__ bo2,
    const _Float16* __restrict__ wf1p, const float* __restrict__ bf1,
    const _Float16* __restrict__ wf2p, const float* __restrict__ bf2,
    float* __restrict__ out) {
  __shared__ __align__(32) _Float16 cbuf[16 * 256];
  __shared__ __align__(32) _Float16 updbuf[16 * 128];
  __shared__ __align__(32) float x2buf[16 * 128];
  __shared__ __align__(32) _Float16 t16[16 * 128];
  __shared__ __align__(32) _Float16 ubuf[16 * 512];
  const int t = threadIdx.x;
  const int lane = t & 31;
  const int w = t >> 5;
  const long e0 = (long)blockIdx.x * 16;
  const int n = w * 16 + (lane & 15);
  const int mh = (lane & 16) ? 8 : 0;

  for (int i = t; i < 2048; i += 256) {
    const int row = i >> 7, col = i & 127;
    cbuf[row * 256 + col] = (_Float16)agg[(e0 + row) * cH + col];
    cbuf[row * 256 + 128 + col] = (_Float16)xdln[(e0 + row) * cH + col];
  }
  __syncthreads();
  {  // gate + upd
    v8f acc;
    const float bb = bg[n];
#pragma unroll
    for (int r = 0; r < 8; ++r) acc[r] = bb;
#pragma unroll
    for (int kc = 0; kc < 8; ++kc)
      acc = wmma_f16(load_a_frag(cbuf, 256, kc * 32),
                     load_b_frag(wgp + (long)w * (8 * 512) + kc * 512), acc);
#pragma unroll
    for (int r = 0; r < 8; ++r) {
      const long e = e0 + mh + r;
      const float g = 1.f / (1.f + expf(-acc[r]));
      const float a0 = agg[e * cH + n];
      const float s0 = sv[e * cH + n];
      updbuf[(mh + r) * cH + n] = (_Float16)(a0 + g * (s0 - a0));
    }
  }
  __syncthreads();
  {  // x2 = xdst + upd Wo^T + Bo
    v8f acc;
    const float bb = bo2[n];
#pragma unroll
    for (int r = 0; r < 8; ++r) acc[r] = bb;
#pragma unroll
    for (int kc = 0; kc < 4; ++kc)
      acc = wmma_f16(load_a_frag(updbuf, cH, kc * 32),
                     load_b_frag(wop2 + (long)w * 2048 + kc * 512), acc);
#pragma unroll
    for (int r = 0; r < 8; ++r)
      x2buf[(mh + r) * cH + n] = xdst[(e0 + mh + r) * cH + n] + acc[r];
  }
  __syncthreads();
  ln_block<16>(x2buf, t16, false, nullptr, 0);
  __syncthreads();
  // FFN up: 128 -> 512 (+relu)
  for (int sub = 0; sub < 4; ++sub) {
    const int nt = w * 4 + sub;
    v8f acc;
    const float bb = bf1[nt * 16 + (lane & 15)];
#pragma unroll
    for (int r = 0; r < 8; ++r) acc[r] = bb;
#pragma unroll
    for (int kc = 0; kc < 4; ++kc)
      acc = wmma_f16(load_a_frag(t16, cH, kc * 32),
                     load_b_frag(wf1p + (long)nt * (4 * 512) + kc * 512), acc);
#pragma unroll
    for (int r = 0; r < 8; ++r)
      ubuf[(mh + r) * 512 + nt * 16 + (lane & 15)] =
          (_Float16)fmaxf(acc[r], 0.f);
  }
  __syncthreads();
  {  // FFN down: 512 -> 128, residual
    v8f acc;
    const float bb = bf2[n];
#pragma unroll
    for (int r = 0; r < 8; ++r) acc[r] = bb;
#pragma unroll
    for (int kc = 0; kc < 16; ++kc)
      acc = wmma_f16(load_a_frag(ubuf, 512, kc * 32),
                     load_b_frag(wf2p + (long)w * (16 * 512) + kc * 512), acc);
#pragma unroll
    for (int r = 0; r < 8; ++r)
      out[(e0 + mh + r) * cH + n] = x2buf[(mh + r) * cH + n] + acc[r];
  }
}

// ---------------- small helpers ----------------
__global__ void xpl0_kernel(const int* __restrict__ plt,
                            const int* __restrict__ pli,
                            const float* __restrict__ eT,
                            const float* __restrict__ eI,
                            float* __restrict__ out) {
  const long i = (long)blockIdx.x * 256 + threadIdx.x;  // over B*PL*H
  const long e = i >> 7;
  const int c = i & 127;
  out[i] = eT[(long)plt[e] * cH + c] + eI[(long)pli[e] * cH + c];
}

__global__ void gather_kernel(const float* __restrict__ xpl,
                              const int* __restrict__ idx,
                              float* __restrict__ out) {
  const long i = (long)blockIdx.x * 256 + threadIdx.x;  // over B*PL*K*H
  const long e = i >> 7;
  const int c = i & 127;
  const long b = e / ((long)cPL * cK);
  const int srcp = idx[e];
  out[i] = xpl[((long)b * cPL + srcp) * cH + c];
}

// ---------------- host ----------------
extern "C" void kernel_launch(void* const* d_in, const int* in_sizes, int n_in,
                              void* d_out, int out_size, void* d_ws,
                              size_t ws_size, hipStream_t stream) {
  (void)in_sizes; (void)n_in; (void)out_size; (void)ws_size;
  const float* magnitude   = (const float*)d_in[0];
  const int*   pt_type     = (const int*)d_in[1];
  const int*   side        = (const int*)d_in[2];
  const int*   mask_pt     = (const int*)d_in[3];
  const int*   pl_type     = (const int*)d_in[4];
  const int*   is_int      = (const int*)d_in[5];
  const float* r_pt2pl     = (const float*)d_in[6];
  const float* r_pl2pl     = (const float*)d_in[7];
  const int*   type_pl2pl  = (const int*)d_in[8];
  const int*   mask_pl2pl  = (const int*)d_in[9];
  const int*   pl2pl_idx   = (const int*)d_in[10];
  const float* emb_pl2pl   = (const float*)d_in[11];
  const float* emb_pt_type = (const float*)d_in[12];
  const float* emb_pt_side = (const float*)d_in[13];
  const float* emb_pl_type = (const float*)d_in[14];
  const float* emb_pl_int  = (const float*)d_in[15];
  const float* fx_freqs = (const float*)d_in[16];
  const float* fx_w1 = (const float*)d_in[17];
  const float* fx_b1 = (const float*)d_in[18];
  const float* fx_w2 = (const float*)d_in[19];
  const float* fx_b2 = (const float*)d_in[20];
  const float* fx_wo = (const float*)d_in[21];
  const float* fx_bo = (const float*)d_in[22];
  const float* fr_freqs = (const float*)d_in[23];
  const float* fr_w1 = (const float*)d_in[24];
  const float* fr_b1 = (const float*)d_in[25];
  const float* fr_w2 = (const float*)d_in[26];
  const float* fr_b2 = (const float*)d_in[27];
  const float* fr_wo = (const float*)d_in[28];
  const float* fr_bo = (const float*)d_in[29];
  const float* wq  = (const float*)d_in[30];
  const float* bq  = (const float*)d_in[31];
  const float* wk  = (const float*)d_in[32];
  const float* wv  = (const float*)d_in[33];
  const float* bv  = (const float*)d_in[34];
  const float* wkr = (const float*)d_in[35];
  const float* wvr = (const float*)d_in[36];
  const float* bvr = (const float*)d_in[37];
  const float* wss = (const float*)d_in[38];
  const float* bss = (const float*)d_in[39];
  const float* wg  = (const float*)d_in[40];
  const float* bg  = (const float*)d_in[41];
  const float* wo_at = (const float*)d_in[42];
  const float* bo_at = (const float*)d_in[43];
  const float* wf1 = (const float*)d_in[44];
  const float* bf1 = (const float*)d_in[45];
  const float* wf2 = (const float*)d_in[46];
  const float* bf2 = (const float*)d_in[47];

  float* out_xpt = (float*)d_out;                  // (B,PL,PT,H)
  float* out_xpl = out_xpt + N_PT * cH;            // (B,PL,H)

  // ---- workspace carve-up (~190 MB total) ----
  char* wsb = (char*)d_ws;
  size_t off = 0;
  auto allocH = [&](size_t nh) {
    _Float16* p = (_Float16*)(wsb + off);
    off += nh * 2; off = (off + 255) & ~(size_t)255;
    return p;
  };
  auto allocF = [&](size_t nf) {
    float* p = (float*)(wsb + off);
    off += nf * 4; off = (off + 255) & ~(size_t)255;
    return p;
  };
  // packed weights
  _Float16* p_fxw1 = allocH(W1H);
  _Float16* p_fxw2 = allocH(W2H);
  _Float16* p_fxwo = allocH(W2H);
  _Float16* p_frw1 = allocH((size_t)6 * W1H);
  _Float16* p_frw2 = allocH((size_t)6 * W2H);
  _Float16* p_frwo = allocH((size_t)2 * W2H);
  _Float16 *p_wq[2], *p_wk[2], *p_wv[2], *p_wkr[2], *p_wvr[2], *p_ws[2],
      *p_wg[2], *p_wo2[2], *p_wf1[2], *p_wf2[2];
  for (int s = 0; s < 2; ++s) {
    p_wq[s] = allocH(W2H);  p_wk[s] = allocH(W2H);  p_wv[s] = allocH(W2H);
    p_wkr[s] = allocH(W2H); p_wvr[s] = allocH(W2H); p_ws[s] = allocH(W2H);
    p_wg[s] = allocH(WGH);  p_wo2[s] = allocH(W2H);
    p_wf1[s] = allocH(WF1H); p_wf2[s] = allocH(WF2H);
  }
  // activations
  float* r_pt  = allocF((size_t)N_PT * cH);
  float* r_pl  = allocF((size_t)N_PLK * cH);
  float* x_pl0 = allocF((size_t)N_PL * cH);
  float* xsrc2 = allocF((size_t)N_PLK * cH);
  float* xdln  = allocF((size_t)N_PL * cH);
  float* qbuf  = allocF((size_t)N_PL * cH);
  float* svbuf = allocF((size_t)N_PL * cH);
  float* aggbf = allocF((size_t)N_PL * cH);
  float* xpl1  = allocF((size_t)N_PL * cH);
  _Float16* k1 = allocH((size_t)N_PT * cH);
  _Float16* v1 = allocH((size_t)N_PT * cH);
  _Float16* k2 = allocH((size_t)N_PLK * cH);
  _Float16* v2 = allocH((size_t)N_PLK * cH);

  auto pack = [&](_Float16* dst, const float* src, int N, int Kdim) {
    const int KC = (Kdim + 31) / 32;
    pack_b_kernel<<<dim3(KC, N / 16), 32, 0, stream>>>(dst, src, N, Kdim, KC);
  };

  // ---- pack all weights ----
  pack(p_fxw1, fx_w1, 128, 129);
  pack(p_fxw2, fx_w2, 128, 128);
  pack(p_fxwo, fx_wo, 128, 128);
  for (int s = 0; s < 2; ++s)
    for (int d = 0; d < 3; ++d) {
      pack(p_frw1 + (size_t)(s * 3 + d) * W1H,
           fr_w1 + (size_t)(s * 3 + d) * 128 * 129, 128, 129);
      pack(p_frw2 + (size_t)(s * 3 + d) * W2H,
           fr_w2 + (size_t)(s * 3 + d) * 128 * 128, 128, 128);
    }
  for (int s = 0; s < 2; ++s)
    pack(p_frwo + (size_t)s * W2H, fr_wo + (size_t)s * 128 * 128, 128, 128);
  for (int s = 0; s < 2; ++s) {
    pack(p_wq[s], wq + (size_t)s * 16384, 128, 128);
    pack(p_wk[s], wk + (size_t)s * 16384, 128, 128);
    pack(p_wv[s], wv + (size_t)s * 16384, 128, 128);
    pack(p_wkr[s], wkr + (size_t)s * 16384, 128, 128);
    pack(p_wvr[s], wvr + (size_t)s * 16384, 128, 128);
    pack(p_ws[s], wss + (size_t)s * 16384, 128, 128);
    pack(p_wg[s], wg + (size_t)s * 128 * 256, 128, 256);
    pack(p_wo2[s], wo_at + (size_t)s * 16384, 128, 128);
    pack(p_wf1[s], wf1 + (size_t)s * 512 * 128, 512, 128);
    pack(p_wf2[s], wf2 + (size_t)s * 128 * 512, 128, 512);
  }

  // ---- embeddings for x_pl ----
  xpl0_kernel<<<(N_PL * cH) / 256, 256, 0, stream>>>(pl_type, is_int,
                                                     emb_pl_type, emb_pl_int,
                                                     x_pl0);

  // ---- Fourier stages ----
  fourier_kernel<<<N_PT / 32, 256, 0, stream>>>(
      magnitude, (long)0, 1, fx_freqs, p_fxw1, fx_b1, p_fxw2, fx_b2, p_fxwo,
      fx_bo, 1, pt_type, side, emb_pt_type, emb_pt_side, out_xpt);
  fourier_kernel<<<N_PT / 32, 256, 0, stream>>>(
      r_pt2pl, N_PT, 3, fr_freqs, p_frw1, fr_b1, p_frw2, fr_b2, p_frwo, fr_bo,
      0, nullptr, nullptr, nullptr, nullptr, r_pt);
  fourier_kernel<<<N_PLK / 32, 256, 0, stream>>>(
      r_pl2pl, N_PLK, 3, fr_freqs + 3 * cF, p_frw1 + (size_t)3 * W1H,
      fr_b1 + 3 * cH, p_frw2 + (size_t)3 * W2H, fr_b2 + 3 * cH, p_frwo + W2H,
      fr_bo + cH, 2, type_pl2pl, nullptr, emb_pl2pl, nullptr, r_pl);

  // ---- attention stage 1: src = x_pt, dst = x_pl0, S = PT ----
  kv_kernel<<<N_PT / 32, 256, 0, stream>>>(out_xpt, r_pt, p_wk[0], p_wkr[0],
                                           p_wv[0], p_wvr[0], bv, bvr, k1, v1);
  qs_kernel<<<N_PL / 16, 256, 0, stream>>>(x_pl0, p_wq[0], bq, p_ws[0], bss,
                                           xdln, qbuf, svbuf);
  attn_core_kernel<<<N_PL, 256, 0, stream>>>(qbuf, k1, v1, mask_pt, cPT, aggbf);
  post_kernel<<<N_PL / 16, 256, 0, stream>>>(
      aggbf, xdln, svbuf, x_pl0, p_wg[0], bg, p_wo2[0], bo_at, p_wf1[0], bf1,
      p_wf2[0], bf2, xpl1);

  // ---- gather x_pl_src from pre-attention x_pl ----
  gather_kernel<<<(N_PLK * cH) / 256, 256, 0, stream>>>(x_pl0, pl2pl_idx,
                                                        xsrc2);

  // ---- attention stage 2: src = gathered, dst = xpl1, S = K ----
  kv_kernel<<<N_PLK / 32, 256, 0, stream>>>(xsrc2, r_pl, p_wk[1], p_wkr[1],
                                            p_wv[1], p_wvr[1], bv + cH,
                                            bvr + cH, k2, v2);
  qs_kernel<<<N_PL / 16, 256, 0, stream>>>(xpl1, p_wq[1], bq + cH, p_ws[1],
                                           bss + cH, xdln, qbuf, svbuf);
  attn_core_kernel<<<N_PL, 256, 0, stream>>>(qbuf, k2, v2, mask_pl2pl, cK,
                                             aggbf);
  post_kernel<<<N_PL / 16, 256, 0, stream>>>(
      aggbf, xdln, svbuf, xpl1, p_wg[1], bg + cH, p_wo2[1], bo_at + cH,
      p_wf1[1], bf1 + 512, p_wf2[1], bf2 + cH, out_xpl);
}
